// BiLSTM_CRF_39943195853037
// MI455X (gfx1250) — matrix operand order repaired and checked by reading, not compiled, not run on hardware
//
#include <hip/hip_runtime.h>
#include <hip/hip_bf16.h>

// ---------------------------------------------------------------------------
// Types for CDNA5 WMMA (wave32): v_wmma_f32_16x16x32_bf16
// ---------------------------------------------------------------------------
typedef __bf16 bf16;
typedef __attribute__((ext_vector_type(8)))  bf16  v8bf;
typedef __attribute__((ext_vector_type(16))) bf16  v16bf;
typedef __attribute__((ext_vector_type(8)))  float v8f;

__device__ __forceinline__ v16bf cat8(v8bf lo, v8bf hi) {
  return __builtin_shufflevector(lo, hi, 0,1,2,3,4,5,6,7,8,9,10,11,12,13,14,15);
}

// A fragment: 16x32 bf16 tile from row-major A [*, lda], top-left (row0, k0).
// ISA layout: lane l (half=l>>4, m=l&15) holds K = k0+half*8+[0..7] (elems 0-7)
// and K = k0+16+half*8+[0..7] (elems 8-15) of row m.
__device__ __forceinline__ v16bf loadA_g(const bf16* __restrict__ A, int lda,
                                         int row0, int k0, int lane) {
  int half = lane >> 4, m = lane & 15;
  const bf16* p = A + (size_t)(row0 + m) * lda + k0 + half * 8;
  return cat8(*(const v8bf*)p, *(const v8bf*)(p + 16));
}

// B fragment: 32x16 tile, supplied as Bt row-major [N, K] (i.e. W as stored).
// ISA layout: lane l (half=l>>4, n=l&15) holds K = k0+half*8+[0..7] (elems 0-7)
// and K = k0+16+half*8+[0..7] (elems 8-15) of col n.
__device__ __forceinline__ v16bf loadB_g(const bf16* __restrict__ Bt, int ldb,
                                         int col0, int k0, int lane) {
  int half = lane >> 4, n = lane & 15;
  const bf16* p = Bt + (size_t)(col0 + n) * ldb + k0 + half * 8;
  return cat8(*(const v8bf*)p, *(const v8bf*)(p + 16));
}

__device__ __forceinline__ float sigmf(float x) { return 1.0f / (1.0f + __expf(-x)); }

// ---------------------------------------------------------------------------
// Problem constants
// ---------------------------------------------------------------------------
constexpr int Bn = 64, Tn = 256, BT = Bn * Tn;
constexpr int CHAR_E = 256, POS_E = 128;
constexpr int HP = 128;                 // pos LSTM hidden per direction
constexpr int HM = 256;                 // main LSTM hidden per direction
constexpr int DM = CHAR_E + POS_E + 2 * HP;  // 640
constexpr int Ln = 20, STARTL = 18, STOPL = 19;

// ---------------------------------------------------------------------------
// f32 -> bf16 convert
// ---------------------------------------------------------------------------
__global__ void cvt_kernel(const float* __restrict__ s, bf16* __restrict__ d, int n) {
  for (int i = blockIdx.x * blockDim.x + threadIdx.x; i < n; i += gridDim.x * blockDim.x)
    d[i] = (bf16)s[i];
}

// Pad W_out [20,512] -> [32,512] bf16 (zero rows 20..31), b_out [20] -> [32] f32.
__global__ void prep_wout(const float* __restrict__ W, const float* __restrict__ b,
                          bf16* __restrict__ Wp, float* __restrict__ bp) {
  int i = blockIdx.x * blockDim.x + threadIdx.x;
  if (i < 32 * 512) {
    int r = i / 512, c = i % 512;
    Wp[i] = (bf16)(r < Ln ? W[r * 512 + c] : 0.0f);
  }
  if (i < 32) bp[i] = (i < Ln) ? b[i] : 0.0f;
}

// ---------------------------------------------------------------------------
// Embedding gather + concat + zero-init of scatter targets
// Xpos  [BT,128]  = tag_pos_emb[tag]
// Xmain [BT,640]  = [char_emb | pos_emb | <pos-LSTM out, zeroed here>]
// Xemit [BT,512]  = zero (filled by main LSTM)
// ---------------------------------------------------------------------------
__global__ void embed_kernel(const int* __restrict__ ci, const int* __restrict__ pi,
                             const int* __restrict__ ti,
                             const float* __restrict__ ce, const float* __restrict__ pe,
                             const float* __restrict__ te,
                             bf16* __restrict__ Xpos, bf16* __restrict__ Xmain,
                             bf16* __restrict__ Xemit) {
  int r = blockIdx.x;
  int c = ci[r], p = pi[r], t = ti[r];
  bf16* xm = Xmain + (size_t)r * DM;
  for (int k = threadIdx.x; k < POS_E; k += blockDim.x)
    Xpos[(size_t)r * POS_E + k] = (bf16)te[(size_t)t * POS_E + k];
  for (int k = threadIdx.x; k < CHAR_E; k += blockDim.x)
    xm[k] = (bf16)ce[(size_t)c * CHAR_E + k];
  for (int k = threadIdx.x; k < POS_E; k += blockDim.x)
    xm[CHAR_E + k] = (bf16)pe[(size_t)p * POS_E + k];
  for (int k = threadIdx.x; k < 2 * HP; k += blockDim.x)
    xm[CHAR_E + POS_E + k] = (bf16)0.0f;
  for (int k = threadIdx.x; k < 2 * HM; k += blockDim.x)
    Xemit[(size_t)r * (2 * HM) + k] = (bf16)0.0f;
}

// ---------------------------------------------------------------------------
// Tiled WMMA GEMM: C[M,N] = A[M,K](bf16) * Bt[N,K]^T(bf16) + bias[N], f32 out.
// Each wave owns a 16 x (16*NT) strip: one A fragment is reused across NT
// B fragments / accumulators per k-chunk (4x less A traffic for NT=4).
// Waves are nt-major so consecutive waves in a block share the same A rows.
// ---------------------------------------------------------------------------
template <int NT>
__global__ void gemm_bias(const bf16* __restrict__ A, const bf16* __restrict__ Bt,
                          const float* __restrict__ bias, float* __restrict__ C,
                          int M, int N, int K) {
  int wave = (blockIdx.x * blockDim.x + threadIdx.x) >> 5;
  int lane = threadIdx.x & 31;
  int ntg = N / (16 * NT);                 // n-strip groups per row-tile
  int total = (M >> 4) * ntg;
  if (wave >= total) wave = total - 1;     // keep EXEC full for WMMA
  int mt = wave / ntg, ng = wave % ntg;
  int n0 = ng * 16 * NT;
  v8f acc[NT];
#pragma unroll
  for (int j = 0; j < NT; ++j) acc[j] = (v8f){};
  for (int k0 = 0; k0 < K; k0 += 32) {
    v16bf a = loadA_g(A, K, mt * 16, k0, lane);
#pragma unroll
    for (int j = 0; j < NT; ++j) {
      v16bf b = loadB_g(Bt, K, n0 + j * 16, k0, lane);
      acc[j] = __builtin_amdgcn_wmma_f32_16x16x32_bf16(false, a, false, b, (short)0,
                                                       acc[j], false, false);
    }
  }
  int half = lane >> 4;
#pragma unroll
  for (int j = 0; j < NT; ++j) {
    int n = n0 + j * 16 + (lane & 15);
    float bv = bias[n];
    float* crow = C + (size_t)(mt * 16 + half * 8) * N + n;
#pragma unroll
    for (int v = 0; v < 8; ++v) crow[(size_t)v * N] = acc[j][v] + bv;
  }
}

// ---------------------------------------------------------------------------
// Recurrent LSTM kernel. One block = 16 batch rows (mt = blockIdx.x).
// Hidden state h [16,H] bf16 in LDS (WMMA A-fragments via ds_load_b128),
// cell state c in f32 registers (D-fragment layout). Each wave owns one
// 16-column hidden tile and computes the 4 gate tiles i/f/g/o per step,
// reusing the A fragment across the four gate WMMAs.
// xW = precomputed x@Wih^T + b (f32, [B*T, 4H]); Whh bf16 [4H, H].
// Forward writes y at t; backward scatters to t_src = len-1-t (masked).
// ---------------------------------------------------------------------------
template <int H, bool BACKWARD>
__global__ void __launch_bounds__(2 * H) lstm_rec(
    const float* __restrict__ xW, const bf16* __restrict__ Whh,
    const int* __restrict__ lengths, bf16* __restrict__ Y, int ldy, int colOff) {
  __shared__ bf16 h_lds[16][H];
  __shared__ int len_l[16];
  const int lane = threadIdx.x & 31;
  const int wv = threadIdx.x >> 5;          // hidden tile index, 0..H/16-1
  const int half = lane >> 4, ln = lane & 15;
  const int bBase = blockIdx.x * 16;
  const int NG = 4 * H;
  const int hcol = wv * 16 + ln;

  for (int i = threadIdx.x; i < 16 * H; i += blockDim.x) (&h_lds[0][0])[i] = (bf16)0.0f;
  if (threadIdx.x < 16) len_l[threadIdx.x] = lengths[bBase + threadIdx.x];
  __syncthreads();

  v8f c = {};
  for (int t = 0; t < Tn; ++t) {
    v8f gi = {}, gf = {}, gg = {}, go = {};
#pragma unroll
    for (int kc = 0; kc < H; kc += 32) {
      const bf16* pa = &h_lds[ln][kc + half * 8];
      v16bf a = cat8(*(const v8bf*)pa, *(const v8bf*)(pa + 16));
      v16bf b0 = loadB_g(Whh, H, 0 * H + wv * 16, kc, lane);
      v16bf b1 = loadB_g(Whh, H, 1 * H + wv * 16, kc, lane);
      v16bf b2 = loadB_g(Whh, H, 2 * H + wv * 16, kc, lane);
      v16bf b3 = loadB_g(Whh, H, 3 * H + wv * 16, kc, lane);
      gi = __builtin_amdgcn_wmma_f32_16x16x32_bf16(false, a, false, b0, (short)0, gi, false, false);
      gf = __builtin_amdgcn_wmma_f32_16x16x32_bf16(false, a, false, b1, (short)0, gf, false, false);
      gg = __builtin_amdgcn_wmma_f32_16x16x32_bf16(false, a, false, b2, (short)0, gg, false, false);
      go = __builtin_amdgcn_wmma_f32_16x16x32_bf16(false, a, false, b3, (short)0, go, false, false);
    }
    __syncthreads();  // all waves done reading h_lds for this step
#pragma unroll
    for (int v = 0; v < 8; ++v) {
      int r = half * 8 + v;
      int b = bBase + r;
      int len = len_l[r];
      bool m = t < len;
      int tx = BACKWARD ? (len - 1 - t) : t;
      tx = tx < 0 ? 0 : (tx > Tn - 1 ? Tn - 1 : tx);
      const float* xr = xW + ((size_t)b * Tn + tx) * (size_t)NG;
      float i_p = gi[v] + xr[0 * H + hcol];
      float f_p = gf[v] + xr[1 * H + hcol];
      float g_p = gg[v] + xr[2 * H + hcol];
      float o_p = go[v] + xr[3 * H + hcol];
      float c2 = sigmf(f_p) * c[v] + sigmf(i_p) * tanhf(g_p);
      float h2 = sigmf(o_p) * tanhf(c2);
      float hold = (float)h_lds[r][hcol];
      c[v] = m ? c2 : c[v];
      float hn = m ? h2 : hold;
      h_lds[r][hcol] = (bf16)hn;
      float y = m ? h2 : 0.0f;
      if (!BACKWARD) {
        Y[((size_t)b * Tn + t) * ldy + colOff + hcol] = (bf16)y;
      } else if (m) {
        Y[((size_t)b * Tn + tx) * ldy + colOff + hcol] = (bf16)y;
      }
    }
    __syncthreads();  // writes visible before next step's fragment reads
  }
}

// ---------------------------------------------------------------------------
// CRF Viterbi decode. Block per batch row (32 threads); emit [BT,32] f32.
// bp buffer: [T][B][32] int. Output tags as float (masked).
// ---------------------------------------------------------------------------
__global__ void viterbi_kernel(const float* __restrict__ emit, const float* __restrict__ trans,
                               const int* __restrict__ lengths, float* __restrict__ out,
                               int* __restrict__ bp) {
  int b = blockIdx.x;
  int lane = threadIdx.x;
  __shared__ float tr[Ln * Ln];
  __shared__ float alpha[Ln];
  __shared__ float nalpha[Ln];
  for (int i = lane; i < Ln * Ln; i += 32) tr[i] = trans[i];
  int len = lengths[b];
  if (lane < Ln) alpha[lane] = tr[STARTL * Ln + lane] + emit[((size_t)b * Tn) * 32 + lane];
  __syncthreads();
  for (int t = 1; t < Tn; ++t) {
    bool m = t < len;
    float best = -3.4e38f;
    int arg = 0;
    if (lane < Ln) {
      for (int i = 0; i < Ln; ++i) {
        float s = alpha[i] + tr[i * Ln + lane];
        if (s > best) { best = s; arg = i; }
      }
    }
    __syncthreads();
    if (lane < Ln) {
      float nv = best + emit[((size_t)b * Tn + t) * 32 + lane];
      nalpha[lane] = m ? nv : alpha[lane];
      bp[((size_t)t * Bn + b) * 32 + lane] = m ? arg : lane;
    }
    __syncthreads();
    if (lane < Ln) alpha[lane] = nalpha[lane];
    __syncthreads();
  }
  if (lane == 0) {
    int tag = 0;
    float bs = -3.4e38f;
    for (int j = 0; j < Ln; ++j) {
      float s = alpha[j] + tr[j * Ln + STOPL];
      if (s > bs) { bs = s; tag = j; }
    }
    for (int t = Tn - 1; t >= 1; --t) {
      out[(size_t)b * Tn + t] = (t < len) ? (float)tag : 0.0f;
      tag = bp[((size_t)t * Bn + b) * 32 + tag];
    }
    out[(size_t)b * Tn] = (0 < len) ? (float)tag : 0.0f;
  }
}

// ---------------------------------------------------------------------------
// Host launcher
// ---------------------------------------------------------------------------
extern "C" void kernel_launch(void* const* d_in, const int* in_sizes, int n_in,
                              void* d_out, int out_size, void* d_ws, size_t ws_size,
                              hipStream_t stream) {
  (void)in_sizes; (void)n_in; (void)out_size; (void)ws_size;
  const int* char_in = (const int*)d_in[0];
  const int* pos_in  = (const int*)d_in[1];
  const int* tag_in  = (const int*)d_in[2];
  const int* lengths = (const int*)d_in[3];
  // d_in[4] = mask (recomputed from lengths; unused)
  const float* char_emb    = (const float*)d_in[5];
  const float* pos_emb     = (const float*)d_in[6];
  const float* tag_pos_emb = (const float*)d_in[7];
  const float* pWih_f = (const float*)d_in[8];
  const float* pWhh_f = (const float*)d_in[9];
  const float* pb_f   = (const float*)d_in[10];
  const float* pWih_b = (const float*)d_in[11];
  const float* pWhh_b = (const float*)d_in[12];
  const float* pb_b   = (const float*)d_in[13];
  const float* mWih_f = (const float*)d_in[14];
  const float* mWhh_f = (const float*)d_in[15];
  const float* mb_f   = (const float*)d_in[16];
  const float* mWih_b = (const float*)d_in[17];
  const float* mWhh_b = (const float*)d_in[18];
  const float* mb_b   = (const float*)d_in[19];
  const float* W_out  = (const float*)d_in[20];
  const float* b_out  = (const float*)d_in[21];
  const float* trans  = (const float*)d_in[22];
  float* out = (float*)d_out;

  // Workspace bump allocator (256B aligned)
  char* wp = (char*)d_ws;
  auto alloc = [&](size_t bytes) -> void* {
    void* p = (void*)wp;
    wp += (bytes + 255) & ~(size_t)255;
    return p;
  };
  bf16* Xpos  = (bf16*)alloc((size_t)BT * POS_E * sizeof(bf16));
  bf16* Xmain = (bf16*)alloc((size_t)BT * DM * sizeof(bf16));
  bf16* Xemit = (bf16*)alloc((size_t)BT * (2 * HM) * sizeof(bf16));
  float* xWpF = (float*)alloc((size_t)BT * 4 * HP * sizeof(float));
  float* xWpB = (float*)alloc((size_t)BT * 4 * HP * sizeof(float));
  float* xWmF = (float*)alloc((size_t)BT * 4 * HM * sizeof(float));
  float* xWmB = (float*)alloc((size_t)BT * 4 * HM * sizeof(float));
  float* emit = (float*)alloc((size_t)BT * 32 * sizeof(float));
  int*   bp   = (int*)alloc((size_t)Tn * Bn * 32 * sizeof(int));
  bf16* wpWihF = (bf16*)alloc((size_t)4 * HP * POS_E * sizeof(bf16));
  bf16* wpWhhF = (bf16*)alloc((size_t)4 * HP * HP * sizeof(bf16));
  bf16* wpWihB = (bf16*)alloc((size_t)4 * HP * POS_E * sizeof(bf16));
  bf16* wpWhhB = (bf16*)alloc((size_t)4 * HP * HP * sizeof(bf16));
  bf16* wmWihF = (bf16*)alloc((size_t)4 * HM * DM * sizeof(bf16));
  bf16* wmWhhF = (bf16*)alloc((size_t)4 * HM * HM * sizeof(bf16));
  bf16* wmWihB = (bf16*)alloc((size_t)4 * HM * DM * sizeof(bf16));
  bf16* wmWhhB = (bf16*)alloc((size_t)4 * HM * HM * sizeof(bf16));
  bf16* wWout  = (bf16*)alloc((size_t)32 * 512 * sizeof(bf16));
  float* bOutP = (float*)alloc(32 * sizeof(float));

  auto conv = [&](const float* s, bf16* d, int n) {
    cvt_kernel<<<(n + 255) / 256, 256, 0, stream>>>(s, d, n);
  };
  conv(pWih_f, wpWihF, 4 * HP * POS_E);
  conv(pWhh_f, wpWhhF, 4 * HP * HP);
  conv(pWih_b, wpWihB, 4 * HP * POS_E);
  conv(pWhh_b, wpWhhB, 4 * HP * HP);
  conv(mWih_f, wmWihF, 4 * HM * DM);
  conv(mWhh_f, wmWhhF, 4 * HM * HM);
  conv(mWih_b, wmWihB, 4 * HM * DM);
  conv(mWhh_b, wmWhhB, 4 * HM * HM);
  prep_wout<<<64, 256, 0, stream>>>(W_out, b_out, wWout, bOutP);

  embed_kernel<<<BT, 256, 0, stream>>>(char_in, pos_in, tag_in, char_emb, pos_emb,
                                       tag_pos_emb, Xpos, Xmain, Xemit);

  // Pos-LSTM input projections: [16384,128] x [128,512], 16x64 strips
  {
    int waves = (BT / 16) * (4 * HP / 64);   // 8192
    gemm_bias<4><<<waves * 32 / 256, 256, 0, stream>>>(Xpos, wpWihF, pb_f, xWpF, BT, 4 * HP, POS_E);
    gemm_bias<4><<<waves * 32 / 256, 256, 0, stream>>>(Xpos, wpWihB, pb_b, xWpB, BT, 4 * HP, POS_E);
  }
  // Pos-LSTM recurrences -> write into Xmain columns [384..511] / [512..639]
  lstm_rec<HP, false><<<4, 2 * HP, 0, stream>>>(xWpF, wpWhhF, lengths, Xmain, DM, CHAR_E + POS_E);
  lstm_rec<HP, true><<<4, 2 * HP, 0, stream>>>(xWpB, wpWhhB, lengths, Xmain, DM, CHAR_E + POS_E + HP);

  // Main-LSTM input projections: [16384,640] x [640,1024], 16x64 strips
  {
    int waves = (BT / 16) * (4 * HM / 64);   // 16384
    gemm_bias<4><<<waves * 32 / 256, 256, 0, stream>>>(Xmain, wmWihF, mb_f, xWmF, BT, 4 * HM, DM);
    gemm_bias<4><<<waves * 32 / 256, 256, 0, stream>>>(Xmain, wmWihB, mb_b, xWmB, BT, 4 * HM, DM);
  }
  // Main-LSTM recurrences -> write into Xemit columns [0..255] / [256..511]
  lstm_rec<HM, false><<<4, 2 * HM, 0, stream>>>(xWmF, wmWhhF, lengths, Xemit, 2 * HM, 0);
  lstm_rec<HM, true><<<4, 2 * HM, 0, stream>>>(xWmB, wmWhhB, lengths, Xemit, 2 * HM, HM);

  // Emit projection: [16384,512] x [512,32] (padded labels), 16x32 strips
  {
    int waves = (BT / 16) * (32 / 32);       // 1024
    gemm_bias<2><<<waves * 32 / 256, 256, 0, stream>>>(Xemit, wWout, bOutP, emit, BT, 32, 2 * HM);
  }

  // CRF Viterbi decode
  viterbi_kernel<<<Bn, 32, 0, stream>>>(emit, trans, lengths, out, bp);
}
